// transformerv1_51694226375222
// MI455X (gfx1250) — matrix operand order repaired and checked by reading, not compiled, not run on hardware
//
#include <hip/hip_runtime.h>
#include <hip/hip_bf16.h>

// ---------------------------------------------------------------------------
// Transformer encoder forward for MI455X (gfx1250, wave32, WMMA bf16,
// async global->LDS staging with ASYNCcnt double buffering).
// B=16, S=1024, D=512, H=4, DK=128, FFN=2048.
// ---------------------------------------------------------------------------

#define S_LEN   1024
#define DMODEL  512
#define NHEAD   4
#define DKH     128
#define BATCH   16
#define TOKENS  (BATCH * S_LEN)   // 16384
#define FFDIM   2048

typedef __bf16 v8bf  __attribute__((ext_vector_type(8)));
typedef __bf16 v16bf __attribute__((ext_vector_type(16)));
typedef float  v8f   __attribute__((ext_vector_type(8)));

#define WMMA_BF16(a, b, c) \
    __builtin_amdgcn_wmma_f32_16x16x32_bf16(false, (a), false, (b), (short)0, (c), false, false)

// ---- gfx1250 async global->LDS path (ASYNCcnt), with compile-safe fallback
#if defined(__has_builtin)
#if __has_builtin(__builtin_amdgcn_global_load_async_to_lds_b128) && \
    __has_builtin(__builtin_amdgcn_s_wait_asynccnt)
#define USE_ASYNC 1
#endif
#endif
#ifndef USE_ASYNC
#define USE_ASYNC 0
#endif

typedef int v4i __attribute__((vector_size(16)));
typedef __attribute__((address_space(1))) v4i* g4p;   // global int4*
typedef __attribute__((address_space(3))) v4i* l4p;   // LDS int4*

__device__ __forceinline__ void async_cp16(const __bf16* g, __bf16* l) {
#if USE_ASYNC
    __builtin_amdgcn_global_load_async_to_lds_b128((g4p)(void*)g, (l4p)(void*)l, 0, 0);
#else
    *(v8bf*)l = *(const v8bf*)g;
#endif
}
#if USE_ASYNC
#define ASYNC_WAIT(n) __builtin_amdgcn_s_wait_asynccnt(n)
#else
#define ASYNC_WAIT(n) ((void)0)
#endif

// A-matrix fragment (16x32 bf16): lane half hf=0 holds K {0..7, 16..23},
// hf=1 holds K {8..15, 24..31} of row M = lane&15.  (ISA 7.12.2)
__device__ __forceinline__ v16bf ld_frag_a(const __bf16* rowp, int hf) {
    v8bf lo = *(const v8bf*)(rowp + hf * 8);
    v8bf hi = *(const v8bf*)(rowp + hf * 8 + 16);
    return __builtin_shufflevector(lo, hi, 0,1,2,3,4,5,6,7,8,9,10,11,12,13,14,15);
}
// B-matrix fragment (32x16 bf16): lane n=lane&15; hf=0 holds K {0..7, 16..23},
// hf=1 holds K {8..15, 24..31}, with LDS stored [n][k].
__device__ __forceinline__ v16bf ld_frag_b(const __bf16* colp, int hf) {
    v8bf lo = *(const v8bf*)(colp + hf * 8);
    v8bf hi = *(const v8bf*)(colp + hf * 8 + 16);
    return __builtin_shufflevector(lo, hi, 0,1,2,3,4,5,6,7,8,9,10,11,12,13,14,15);
}

// ---------------------------------------------------------------------------
// Pre-passes: fp32 -> bf16 convert, and weight transpose+convert ([K,N]->[N,K])
// ---------------------------------------------------------------------------
__global__ __launch_bounds__(256) void cvt_bf16_kernel(
    const float* __restrict__ in, __bf16* __restrict__ out)
{
    size_t i = ((size_t)blockIdx.x * 256 + threadIdx.x) * 8;
    float4 a = *(const float4*)(in + i);
    float4 b = *(const float4*)(in + i + 4);
    v8bf o = { (__bf16)a.x, (__bf16)a.y, (__bf16)a.z, (__bf16)a.w,
               (__bf16)b.x, (__bf16)b.y, (__bf16)b.z, (__bf16)b.w };
    *(v8bf*)(out + i) = o;
}

__global__ __launch_bounds__(256) void transpose_bf16_kernel(
    const float* __restrict__ in, __bf16* __restrict__ out, int K, int N)
{
    __shared__ float t[32][33];
    const float* ip = in + (size_t)blockIdx.z * K * N;
    __bf16* op = out + (size_t)blockIdx.z * N * K;
    int tx = threadIdx.x & 31, ty = threadIdx.x >> 5;
    int kb = blockIdx.y * 32, nb = blockIdx.x * 32;
#pragma unroll
    for (int i = 0; i < 4; ++i)
        t[ty + i * 8][tx] = ip[(size_t)(kb + ty + i * 8) * N + nb + tx];
    __syncthreads();
#pragma unroll
    for (int i = 0; i < 4; ++i)
        op[(size_t)(nb + ty + i * 8) * K + kb + tx] = (__bf16)t[tx][ty + i * 8];
}

// ---------------------------------------------------------------------------
// Generic tiled GEMM: C[M,N] = A[M,K] @ Bt[N,K]^T (+epilogue), all-bf16 inputs.
// Block tile 128x128, K-step 32, 256 threads = 8 waves (2x4), wave tile 64x32.
// Double-buffered LDS, async global->LDS staging.
// EPI: 0=plain f32, 1=bias f32, 2=bias+relu bf16, 3=bias+posenc dual f32+bf16,
//      4=QKV scatter bf16 (fused N=1536, Q scaled by 1/sqrt(DK))
// ---------------------------------------------------------------------------
template <int EPI>
__global__ __launch_bounds__(256) void gemm2_kernel(
    const __bf16* __restrict__ A, const __bf16* __restrict__ Bt,
    const float* __restrict__ bias,
    float* __restrict__ Cf, __bf16* __restrict__ Cb,
    int N, int K)
{
    constexpr int LDT = 40;  // 32 + 8 pad; 80B row stride (16B multiple)
    __shared__ __bf16 Al[2][128 * LDT];
    __shared__ __bf16 Bl[2][128 * LDT];

    const int tid  = threadIdx.x;
    const int wave = tid >> 5, lane = tid & 31;
    const int r = lane & 15, hf = lane >> 4;
    const int wm = wave >> 2, wn = wave & 3;            // 2 x 4 wave grid
    const int mb = blockIdx.x * 128, nb = blockIdx.y * 128;

    auto stage = [&](int it, int buf) {
        const int k0 = it * 32;
#pragma unroll
        for (int i = 0; i < 2; ++i) {
            int idx = tid + i * 256;
            int row = idx >> 2, kc = (idx & 3) * 8;
            async_cp16(A  + (size_t)(mb + row) * K + k0 + kc, &Al[buf][row * LDT + kc]);
            async_cp16(Bt + (size_t)(nb + row) * K + k0 + kc, &Bl[buf][row * LDT + kc]);
        }
    };

    v8f acc[4][2] = {};
    const int NK = K / 32;
    stage(0, 0);
    for (int it = 0; it < NK; ++it) {
        const int cur = it & 1;
        if (it + 1 < NK) {
            stage(it + 1, cur ^ 1);
            ASYNC_WAIT(4);   // keep only next tile's 4 ops in flight
        } else {
            ASYNC_WAIT(0);
        }
        __syncthreads();

        v16bf af[4], bf2[2];
#pragma unroll
        for (int mt = 0; mt < 4; ++mt)
            af[mt] = ld_frag_a(&Al[cur][(wm * 64 + mt * 16 + r) * LDT], hf);
#pragma unroll
        for (int nt = 0; nt < 2; ++nt)
            bf2[nt] = ld_frag_b(&Bl[cur][(wn * 32 + nt * 16 + r) * LDT], hf);
#pragma unroll
        for (int mt = 0; mt < 4; ++mt)
#pragma unroll
            for (int nt = 0; nt < 2; ++nt)
                acc[mt][nt] = WMMA_BF16(af[mt], bf2[nt], acc[mt][nt]);
        __syncthreads();
    }

    // epilogue (C/D layout: lane holds col n=r, rows hf*8+j)
#pragma unroll
    for (int mt = 0; mt < 4; ++mt) {
#pragma unroll
        for (int nt = 0; nt < 2; ++nt) {
            int col  = nb + wn * 32 + nt * 16 + r;
            int row0 = mb + wm * 64 + mt * 16 + hf * 8;
            float bv = (EPI >= 1 && EPI <= 3) ? bias[col] : 0.0f;
#pragma unroll
            for (int j = 0; j < 8; ++j) {
                int row = row0 + j;
                float v = acc[mt][nt][j] + bv;
                if constexpr (EPI == 0) {
                    Cf[(size_t)row * N + col] = v;
                } else if constexpr (EPI == 1) {
                    Cf[(size_t)row * N + col] = v;
                } else if constexpr (EPI == 2) {
                    Cb[(size_t)row * N + col] = (__bf16)fmaxf(v, 0.0f);
                } else if constexpr (EPI == 3) {
                    int s = row & (S_LEN - 1);
                    float twoi = (float)(col & ~1);
                    float ang  = (float)s * __expf(twoi * -0.017988945257766374f); // -ln(1e4)/512
                    v += (col & 1) ? __cosf(ang) : __sinf(ang);
                    Cf[(size_t)row * N + col] = v;
                    Cb[(size_t)row * N + col] = (__bf16)v;
                } else {  // EPI == 4: QKV scatter, layout [type][B][H][S][DK]
                    int type = col >> 9, rest = col & 511;
                    int hd = rest >> 7, e = rest & 127;
                    int b = row >> 10, s = row & (S_LEN - 1);
                    float sc = (type == 0) ? 0.08838834764831845f : 1.0f; // 1/sqrt(128)
                    size_t off = (size_t)type * (BATCH * NHEAD * S_LEN * DKH)
                               + ((size_t)(b * NHEAD + hd) << 17)
                               + (size_t)s * DKH + e;
                    Cb[off] = (__bf16)(v * sc);
                }
            }
        }
    }
}

// ---------------------------------------------------------------------------
// Attention: per (b,h) and 32-query block; exact two-pass softmax with full
// 32x1024 fp32 score strip in LDS; async double-buffered K tiles.
// ---------------------------------------------------------------------------
#define ATTN_LDS_BYTES 219264

__global__ __launch_bounds__(256) void attn_kernel(
    const __bf16* __restrict__ Qg, const __bf16* __restrict__ Kg,
    const __bf16* __restrict__ Vg, __bf16* __restrict__ Og)
{
    const int tid = threadIdx.x;
    const int wave = tid >> 5, lane = tid & 31;
    const int r = lane & 15, hf = lane >> 4;
    const int q0 = blockIdx.x * 32;
    const size_t base = (size_t)blockIdx.y * (S_LEN * DKH);
    const __bf16* Qp = Qg + base + (size_t)q0 * DKH;
    const __bf16* Kp = Kg + base;
    const __bf16* Vp = Vg + base;
    __bf16* Op = Og + base + (size_t)q0 * DKH;

    extern __shared__ char smem[];
    float*  sc  = (float*)smem;                      // [32][1032] fp32 scores
    __bf16* qt  = (__bf16*)(smem + 132096);          // [32][136]
    __bf16* kt0 = qt + 32 * 136;                     // [2][128][136] K tiles / V^T
    __bf16* pt  = kt0 + 2 * 128 * 136;               // [32][136] P tile bf16
    float*  rowinv = (float*)(smem + 219136);        // [32]

    auto kstage = [&](int t, int buf) {
        __bf16* kb = kt0 + buf * (128 * 136);
#pragma unroll
        for (int i = 0; i < 8; ++i) {
            int idx = tid + i * 256;
            int row = idx >> 4, e = (idx & 15) * 8;
            async_cp16(Kp + (size_t)(t * 128 + row) * DKH + e, &kb[row * 136 + e]);
        }
    };

    // stage Q block + first K tile
#pragma unroll
    for (int i = 0; i < 2; ++i) {
        int idx = tid + i * 256;
        int row = idx >> 4, e = (idx & 15) * 8;
        async_cp16(Qp + (size_t)row * DKH + e, &qt[row * 136 + e]);
    }
    kstage(0, 0);

    // ---- pass 1: scores = (Q*scale) @ K^T into LDS strip
    for (int t = 0; t < 8; ++t) {
        if (t + 1 < 8) {
            kstage(t + 1, (t + 1) & 1);
            ASYNC_WAIT(8);
        } else {
            ASYNC_WAIT(0);
        }
        __syncthreads();

        const __bf16* kb = kt0 + (t & 1) * (128 * 136);
        v8f s0 = {}, s1 = {};
#pragma unroll
        for (int ks = 0; ks < 4; ++ks) {
            v16bf bb = ld_frag_b(&kb[(wave * 16 + r) * 136 + ks * 32], hf);
            v16bf a0 = ld_frag_a(&qt[r * 136 + ks * 32], hf);
            v16bf a1 = ld_frag_a(&qt[(16 + r) * 136 + ks * 32], hf);
            s0 = WMMA_BF16(a0, bb, s0);
            s1 = WMMA_BF16(a1, bb, s1);
        }
        int col = t * 128 + wave * 16 + r;
#pragma unroll
        for (int j = 0; j < 8; ++j) {
            sc[(hf * 8 + j) * 1032 + col]      = s0[j];
            sc[(16 + hf * 8 + j) * 1032 + col] = s1[j];
        }
        __syncthreads();
    }

    // ---- softmax over each 1024-wide row (8 threads/row, shfl butterfly)
    {
        int row = tid >> 3, sub = tid & 7;
        float* pr = sc + row * 1032 + sub * 128;
        float mx = -3.0e38f;
#pragma unroll 4
        for (int i = 0; i < 32; ++i) {
            float4 v = *(const float4*)(pr + i * 4);
            mx = fmaxf(mx, fmaxf(fmaxf(v.x, v.y), fmaxf(v.z, v.w)));
        }
        mx = fmaxf(mx, __shfl_xor(mx, 1));
        mx = fmaxf(mx, __shfl_xor(mx, 2));
        mx = fmaxf(mx, __shfl_xor(mx, 4));
        float sum = 0.0f;
#pragma unroll 4
        for (int i = 0; i < 32; ++i) {
            float4 v = *(float4*)(pr + i * 4);
            v.x = __expf(v.x - mx); v.y = __expf(v.y - mx);
            v.z = __expf(v.z - mx); v.w = __expf(v.w - mx);
            *(float4*)(pr + i * 4) = v;
            sum += v.x + v.y + v.z + v.w;
        }
        sum += __shfl_xor(sum, 1);
        sum += __shfl_xor(sum, 2);
        sum += __shfl_xor(sum, 4);
        if (sub == 0) rowinv[row] = 1.0f / sum;
    }
    __syncthreads();

    // ---- pass 2: O = P @ V   (V transposed into kt buffer 0)
    v8f o0 = {}, o1 = {};
    for (int t = 0; t < 8; ++t) {
#pragma unroll
        for (int i = 0; i < 8; ++i) {
            int idx = tid + i * 256;
            int kv = idx >> 4, e0 = (idx & 15) * 8;
            v8bf vv = *(const v8bf*)(Vp + (size_t)(t * 128 + kv) * DKH + e0);
#pragma unroll
            for (int j = 0; j < 8; ++j) kt0[(e0 + j) * 136 + kv] = vv[j];
        }
#pragma unroll
        for (int i = 0; i < 16; ++i) {
            int idx = tid + i * 256;
            int m = idx >> 7, kv = idx & 127;
            pt[m * 136 + kv] = (__bf16)(sc[m * 1032 + t * 128 + kv] * rowinv[m]);
        }
        __syncthreads();
#pragma unroll
        for (int ks = 0; ks < 4; ++ks) {
            v16bf bb = ld_frag_b(&kt0[(wave * 16 + r) * 136 + ks * 32], hf);
            v16bf a0 = ld_frag_a(&pt[r * 136 + ks * 32], hf);
            v16bf a1 = ld_frag_a(&pt[(16 + r) * 136 + ks * 32], hf);
            o0 = WMMA_BF16(a0, bb, o0);
            o1 = WMMA_BF16(a1, bb, o1);
        }
        __syncthreads();
    }

    int col = wave * 16 + r;
#pragma unroll
    for (int j = 0; j < 8; ++j) {
        Op[(size_t)(hf * 8 + j) * DKH + col]      = (__bf16)o0[j];
        Op[(size_t)(16 + hf * 8 + j) * DKH + col] = (__bf16)o1[j];
    }
}

// ---------------------------------------------------------------------------
// Residual add + LayerNorm: one wave per 512-row; fp32 + bf16 shadow outputs
// ---------------------------------------------------------------------------
__global__ __launch_bounds__(256) void ln_kernel(
    const float* __restrict__ A, const float* __restrict__ Bres,
    const float* __restrict__ g, const float* __restrict__ bb,
    float* __restrict__ Out, __bf16* __restrict__ Outb)
{
    int row  = blockIdx.x * 8 + (threadIdx.x >> 5);
    int lane = threadIdx.x & 31;
    size_t ro = (size_t)row * DMODEL + lane * 16;

    float v[16];
    float s = 0.0f, s2 = 0.0f;
#pragma unroll
    for (int i = 0; i < 4; ++i) {
        float4 a = *(const float4*)(A + ro + i * 4);
        float4 b = *(const float4*)(Bres + ro + i * 4);
        v[i * 4 + 0] = a.x + b.x; v[i * 4 + 1] = a.y + b.y;
        v[i * 4 + 2] = a.z + b.z; v[i * 4 + 3] = a.w + b.w;
#pragma unroll
        for (int j = 0; j < 4; ++j) { float t = v[i * 4 + j]; s += t; s2 += t * t; }
    }
#pragma unroll
    for (int m = 16; m >= 1; m >>= 1) { s += __shfl_xor(s, m); s2 += __shfl_xor(s2, m); }
    float mu  = s * (1.0f / DMODEL);
    float var = s2 * (1.0f / DMODEL) - mu * mu;
    float rs  = rsqrtf(var + 1e-5f);
#pragma unroll
    for (int j = 0; j < 16; ++j) {
        int c = lane * 16 + j;
        float o = (v[j] - mu) * rs * g[c] + bb[c];
        Out[(size_t)row * DMODEL + c] = o;
        if (Outb) Outb[(size_t)row * DMODEL + c] = (__bf16)o;
    }
}

// ---------------------------------------------------------------------------
// MaxPool(16) over features + FC[32768,2] + log_softmax; 1 block per batch
// ---------------------------------------------------------------------------
__global__ __launch_bounds__(256) void head_kernel(
    const float* __restrict__ h2, const float* __restrict__ fw,
    const float* __restrict__ fb, float* __restrict__ out)
{
    __shared__ float r0[256], r1[256];
    int b = blockIdx.x, tid = threadIdx.x;
    float a0 = 0.0f, a1 = 0.0f;
    for (int f = tid; f < S_LEN * (DMODEL / 16); f += 256) {
        int s = f >> 5, c = f & 31;
        const float* p = h2 + ((size_t)b * S_LEN + s) * DMODEL + c * 16;
        float m = -3.0e38f;
#pragma unroll
        for (int i = 0; i < 4; ++i) {
            float4 x = *(const float4*)(p + i * 4);
            m = fmaxf(m, fmaxf(fmaxf(x.x, x.y), fmaxf(x.z, x.w)));
        }
        a0 += m * fw[f * 2 + 0];
        a1 += m * fw[f * 2 + 1];
    }
    r0[tid] = a0; r1[tid] = a1;
    __syncthreads();
    for (int st = 128; st > 0; st >>= 1) {
        if (tid < st) { r0[tid] += r0[tid + st]; r1[tid] += r1[tid + st]; }
        __syncthreads();
    }
    if (tid == 0) {
        float l0 = r0[0] + fb[0], l1 = r1[0] + fb[1];
        float mx  = fmaxf(l0, l1);
        float lse = mx + logf(__expf(l0 - mx) + __expf(l1 - mx));
        out[b * 2 + 0] = l0 - lse;
        out[b * 2 + 1] = l1 - lse;
    }
}

// ---------------------------------------------------------------------------
// Host-side orchestration
// ---------------------------------------------------------------------------
extern "C" void kernel_launch(void* const* d_in, const int* in_sizes, int n_in,
                              void* d_out, int out_size, void* d_ws, size_t ws_size,
                              hipStream_t stream) {
    (void)in_sizes; (void)n_in; (void)out_size; (void)ws_size;

    const float* x     = (const float*)d_in[0];
    const float* emb_w = (const float*)d_in[1];
    const float* emb_b = (const float*)d_in[2];
    const float* WQ    = (const float*)d_in[3];
    const float* WK    = (const float*)d_in[4];
    const float* WV    = (const float*)d_in[5];
    const float* WO    = (const float*)d_in[6];
    const float* g1    = (const float*)d_in[7];
    const float* b1    = (const float*)d_in[8];
    const float* l1w   = (const float*)d_in[9];
    const float* l1b   = (const float*)d_in[10];
    const float* l2w   = (const float*)d_in[11];
    const float* l2b   = (const float*)d_in[12];
    const float* g2    = (const float*)d_in[13];
    const float* b2    = (const float*)d_in[14];
    const float* fcw   = (const float*)d_in[15];
    const float* fcb   = (const float*)d_in[16];

    char* ws = (char*)d_ws;
    float*  h    = (float*)(ws + 0);                      // 33.5 MB
    __bf16* hb   = (__bf16*)(ws + 33554432);              // 16.8 MB
    __bf16* xb   = (__bf16*)(ws + 50331648);              // 16.8 MB
    __bf16* qkvb = (__bf16*)(ws + 67108864);              // 50.3 MB [3][B][H][S][DK]
    __bf16* aob  = (__bf16*)(ws + 117440512);             // 16.8 MB [B][H][S][DK]
    float*  wo   = (float*)(ws + 134217728);              // 33.5 MB
    float*  h1   = (float*)(ws + 167772160);              // 33.5 MB
    __bf16* h1b  = (__bf16*)(ws + 201326592);             // 16.8 MB
    __bf16* mid  = (__bf16*)(ws + 218103808);             // 67.1 MB
    float*  f2   = (float*)(ws + 285212672);              // 33.5 MB
    float*  h2   = (float*)(ws + 318767104);              // 33.5 MB
    __bf16* embT = (__bf16*)(ws + 352321536);             // [512][512]
    __bf16* woT  = (__bf16*)(ws + 352845824);             // [512][512]
    __bf16* l1T  = (__bf16*)(ws + 353370112);             // [2048][512]
    __bf16* l2T  = (__bf16*)(ws + 355467264);             // [512][2048]
    __bf16* qkvT = (__bf16*)(ws + 357564416);             // [12][128][512]

    static int attr_set = 0;
    if (!attr_set) {
        (void)hipFuncSetAttribute(reinterpret_cast<const void*>(attn_kernel),
                                  hipFuncAttributeMaxDynamicSharedMemorySize, ATTN_LDS_BYTES);
        attr_set = 1;
    }

    const int QKV_ELEMS = BATCH * NHEAD * S_LEN * DKH;  // per-tensor elems

    // 0) operand preparation: bf16 x, transposed bf16 weights
    cvt_bf16_kernel<<<(TOKENS * DMODEL) / (256 * 8), 256, 0, stream>>>(x, xb);
    transpose_bf16_kernel<<<dim3(16, 16, 1), 256, 0, stream>>>(emb_w, embT, DMODEL, DMODEL);
    transpose_bf16_kernel<<<dim3(16, 16, 1), 256, 0, stream>>>(WO, woT, DMODEL, DMODEL);
    transpose_bf16_kernel<<<dim3(64, 16, 1), 256, 0, stream>>>(l1w, l1T, DMODEL, FFDIM);
    transpose_bf16_kernel<<<dim3(16, 64, 1), 256, 0, stream>>>(l2w, l2T, FFDIM, DMODEL);
    transpose_bf16_kernel<<<dim3(4, 16, 4), 256, 0, stream>>>(WQ, qkvT + 0 * 4 * DKH * DMODEL, DMODEL, DKH);
    transpose_bf16_kernel<<<dim3(4, 16, 4), 256, 0, stream>>>(WK, qkvT + 1 * 4 * DKH * DMODEL, DMODEL, DKH);
    transpose_bf16_kernel<<<dim3(4, 16, 4), 256, 0, stream>>>(WV, qkvT + 2 * 4 * DKH * DMODEL, DMODEL, DKH);

    // 1) h = x @ emb_w + emb_b + pos_encoding   (dual fp32 + bf16 out)
    gemm2_kernel<3><<<dim3(TOKENS / 128, DMODEL / 128), 256, 0, stream>>>(
        xb, embT, emb_b, h, hb, DMODEL, DMODEL);

    // 2) fused QKV: [16384,512] @ [512,1536], scatter into [3][B][H][S][DK]
    gemm2_kernel<4><<<dim3(TOKENS / 128, 12), 256, 0, stream>>>(
        hb, qkvT, nullptr, nullptr, qkvb, 3 * DMODEL, DMODEL);

    // 3) attention -> aob in [B][H][S][DK] (raw reshape to [B,S,H*DK] is free)
    attn_kernel<<<dim3(S_LEN / 32, BATCH * NHEAD), 256, ATTN_LDS_BYTES, stream>>>(
        qkvb, qkvb + QKV_ELEMS, qkvb + 2 * QKV_ELEMS, aob);

    // 4) wo = ao_flat @ WO
    gemm2_kernel<0><<<dim3(TOKENS / 128, DMODEL / 128), 256, 0, stream>>>(
        aob, woT, nullptr, wo, nullptr, DMODEL, DMODEL);

    // 5) h1 = LayerNorm(wo + h; g1, b1)  (+ bf16 shadow)
    ln_kernel<<<TOKENS / 8, 256, 0, stream>>>(wo, h, g1, b1, h1, h1b);

    // 6) mid = relu(h1 @ l1_w + l1_b)  (bf16)
    gemm2_kernel<2><<<dim3(TOKENS / 128, FFDIM / 128), 256, 0, stream>>>(
        h1b, l1T, l1b, nullptr, mid, FFDIM, DMODEL);

    // 7) f2 = mid @ l2_w + l2_b
    gemm2_kernel<1><<<dim3(TOKENS / 128, DMODEL / 128), 256, 0, stream>>>(
        mid, l2T, l2b, f2, nullptr, DMODEL, FFDIM);

    // 8) h2 = LayerNorm(f2 + h1; g2, b2)
    ln_kernel<<<TOKENS / 8, 256, 0, stream>>>(f2, h1, g2, b2, h2, nullptr);

    // 9) maxpool16 + fc + log_softmax
    head_kernel<<<BATCH, 256, 0, stream>>>(h2, fcw, fcb, (float*)d_out);
}
